// MultiHeadAttentionBlock_89730456748578
// MI455X (gfx1250) — matrix operand order repaired and checked by reading, not compiled, not run on hardware
//
#include <hip/hip_runtime.h>

// ---------------------------------------------------------------------------
// MQA attention block for MI455X (gfx1250): f16 WMMA + fp32 accumulate,
// flash-attention streaming (no [B,H,S,S] score materialization).
//
//   q = query @ Wq + bq ; k = key @ Wk + bk ; v = value @ Wv + bv
//   O = softmax(mask ? qk^T/sqrt(dk) : -1e9) v ; out = O @ Wo + bo
//
// 128-thread blocks = 4 wave32 waves; each wave owns a 16x64 strip of the
// 64x64 block tile (4x v_wmma_f32_16x16x32_f16 accumulators). All fragment
// feeds are contiguous per lane -> every fragment is 2x ds_load_b128.
// Staging loops are fixed-trip and split load/store so 4-16 global loads are
// in flight before the first wait (no exec-mask guards, full MLP).
// ---------------------------------------------------------------------------

#define H_   16
#define DK_  64
#define DM_  1024
#define B_   2
#define S_   2048

typedef __attribute__((ext_vector_type(16))) _Float16 v16h;
typedef __attribute__((ext_vector_type(8)))  _Float16 v8h;
typedef __attribute__((ext_vector_type(4)))  _Float16 v4h;
typedef __attribute__((ext_vector_type(8)))  float    v8f;

// --- WMMA 16x16x32 f16 fragment loaders (wave32) ---------------------------
// A 16x32 (MxK), row m = lane&15; element e -> K = e + (e>=8?8:0) + (hi?8:0)
// => two contiguous 8-half (16B) runs at K = kh and K = kh+16.
__device__ __forceinline__ v16h load_frag_a(const _Float16* p, int ld) {
  const int lane = threadIdx.x & 31;
  const int m  = lane & 15;
  const int kh = (lane >> 4) << 3;
  const v8h lo = *(const v8h*)&p[m * ld + kh];
  const v8h hi = *(const v8h*)&p[m * ld + kh + 16];
  return __builtin_shufflevector(lo, hi, 0, 1, 2, 3, 4, 5, 6, 7,
                                 8, 9, 10, 11, 12, 13, 14, 15);
}

// B 32x16 (KxN) from K-transposed storage p[n*ld + k]; col n = lane&15,
// element e -> K = e + (e>=8?8:0) + (hi?8:0)  => two 8-half runs at K = kb, kb+16.
__device__ __forceinline__ v16h load_frag_bt(const _Float16* p, int ld) {
  const int lane = threadIdx.x & 31;
  const int n  = lane & 15;
  const int kb = (lane >> 4) << 3;
  const v8h lo = *(const v8h*)&p[n * ld + kb];
  const v8h hi = *(const v8h*)&p[n * ld + kb + 16];
  return __builtin_shufflevector(lo, hi, 0, 1, 2, 3, 4, 5, 6, 7,
                                 8, 9, 10, 11, 12, 13, 14, 15);
}

__device__ __forceinline__ v4h cvt4(const float4 f) {
  v4h h;
  h[0] = (_Float16)f.x; h[1] = (_Float16)f.y;
  h[2] = (_Float16)f.z; h[3] = (_Float16)f.w;
  return h;
}

// --- Generic GEMM: C[MxN] = A[MxK] @ B[KxN] + bias, fp32 io, f16 WMMA ------
#define AS_LD 40   // 80B rows: 16B-aligned (b128), 20-dword stride, no conflicts
#define BT_LD 40   // same for K-transposed B tile [n][k]

__global__ __launch_bounds__(128)
void gemm_bias_f16wmma(const float* __restrict__ A, const float* __restrict__ Bm,
                       const float* __restrict__ bias, float* __restrict__ C,
                       int M, int N, int K) {
  __shared__ alignas(16) _Float16 As[64][AS_LD];    // [m][k] 64x32 used
  __shared__ alignas(16) _Float16 BsT[64][BT_LD];   // [n][k] 64x32 used
  const int bn = blockIdx.x * 64;
  const int bm = blockIdx.y * 64;
  const int tid  = threadIdx.x;
  const int lane = tid & 31;
  const int wave = tid >> 5;
  v8f acc[4] = {};
  for (int k0 = 0; k0 < K; k0 += 32) {
    // A tile 64x32: 4 float4 per thread, batched loads then stores
    {
      float4 fa[4];
#pragma unroll
      for (int j = 0; j < 4; ++j) {
        const int i = tid + j * 128;
        const int r = i >> 3, c = (i & 7) << 2;
        fa[j] = *(const float4*)&A[(size_t)(bm + r) * K + (k0 + c)];
      }
#pragma unroll
      for (int j = 0; j < 4; ++j) {
        const int i = tid + j * 128;
        const int r = i >> 3, c = (i & 7) << 2;
        *(v4h*)&As[r][c] = cvt4(fa[j]);
      }
    }
    // B tile 32x64 -> K-transposed [n][k]: 4-deep K columns per thread,
    // coalesced across consecutive-n threads; batched loads then b64 stores
    {
      float fb[16];
#pragma unroll
      for (int jq = 0; jq < 4; ++jq) {
        const int i = tid + jq * 128;
        const int n = i & 63, k4 = (i >> 6) << 2;
#pragma unroll
        for (int j = 0; j < 4; ++j)
          fb[jq * 4 + j] = Bm[(size_t)(k0 + k4 + j) * N + (bn + n)];
      }
#pragma unroll
      for (int jq = 0; jq < 4; ++jq) {
        const int i = tid + jq * 128;
        const int n = i & 63, k4 = (i >> 6) << 2;
        v4h hv;
#pragma unroll
        for (int j = 0; j < 4; ++j) hv[j] = (_Float16)fb[jq * 4 + j];
        *(v4h*)&BsT[n][k4] = hv;
      }
    }
    // prefetch next K-chunk toward L2 while this chunk computes
    if (k0 + 32 < K) {
      __builtin_prefetch(&A[(size_t)(bm + (tid >> 1)) * K + (k0 + 32) + (tid & 1) * 16], 0, 0);
      __builtin_prefetch(&Bm[(size_t)(k0 + 32 + (tid >> 2)) * N + bn + (tid & 3) * 16], 0, 0);
    }
    __syncthreads();
    const v16h af = load_frag_a(&As[wave * 16][0], AS_LD);
#pragma unroll
    for (int t = 0; t < 4; ++t) {
      const v16h bf = load_frag_bt(&BsT[t * 16][0], BT_LD);
      acc[t] = __builtin_amdgcn_wmma_f32_16x16x32_f16(
          false, af, false, bf, (short)0, acc[t], false, false);
    }
    __syncthreads();
  }
  const int n0 = lane & 15;
  const int mh = (lane >> 4) << 3;
#pragma unroll
  for (int t = 0; t < 4; ++t)
#pragma unroll
    for (int e = 0; e < 8; ++e) {
      const int m = bm + wave * 16 + mh + e;
      const int n = bn + t * 16 + n0;
      C[(size_t)m * N + n] = acc[t][e] + bias[n];
    }
}

// --- Flash MQA: per block one (b,h) and 64 query rows ----------------------
#define T_LD 72    // 144B rows: 16B-aligned (b128), 36-dword stride, no conflicts

__global__ __launch_bounds__(128)
void flash_mqa(const float* __restrict__ q, const float* __restrict__ k,
               const float* __restrict__ v, const int* __restrict__ mask,
               float* __restrict__ x) {
  __shared__ alignas(16) _Float16 Qs [64][T_LD];   // [qrow][dk]
  __shared__ alignas(16) _Float16 Ks [64][T_LD];   // [key][dk]  (B^T for QK^T)
  __shared__ alignas(16) _Float16 VsT[64][T_LD];   // [dk][key]  (B^T for PV)
  __shared__ alignas(16) _Float16 Ps [64][T_LD];   // [qrow][key]
  __shared__ alignas(4) unsigned char Ms[64][68];  // mask tile, packed bytes
  const int qb = blockIdx.x * 64;
  const int bh = blockIdx.y;
  const int b  = bh >> 4;    // / H_
  const int h  = bh & 15;    // % H_
  const int tid  = threadIdx.x;
  const int lane = tid & 31;
  const int wave = tid >> 5;
  const size_t qbase = (size_t)b * S_ * DM_ + (size_t)h * DK_;
  const size_t kvb   = (size_t)b * S_ * DK_;
  const size_t mb    = (size_t)b * S_ * S_;

  // Q tile, pre-scaled by 1/sqrt(dk) = 0.125 (8 float4/thread, 2 batches of 4)
#pragma unroll
  for (int half = 0; half < 2; ++half) {
    float4 fq[4];
#pragma unroll
    for (int j = 0; j < 4; ++j) {
      const int i = tid + (half * 4 + j) * 128;
      const int r = i >> 4, c = (i & 15) << 2;
      fq[j] = *(const float4*)&q[qbase + (size_t)(qb + r) * DM_ + c];
    }
#pragma unroll
    for (int j = 0; j < 4; ++j) {
      const int i = tid + (half * 4 + j) * 128;
      const int r = i >> 4, c = (i & 15) << 2;
      float4 f = fq[j];
      f.x *= 0.125f; f.y *= 0.125f; f.z *= 0.125f; f.w *= 0.125f;
      *(v4h*)&Qs[r][c] = cvt4(f);
    }
  }

  v8f o[4] = {};
  float mrun[8], lrun[8];
#pragma unroll
  for (int e = 0; e < 8; ++e) { mrun[e] = -1e30f; lrun[e] = 0.0f; }

  const int n0 = lane & 15;
  const int mh = (lane >> 4) << 3;

  for (int kb = 0; kb < S_; kb += 64) {
    // K tile (row-major = already B^T), batched
#pragma unroll
    for (int half = 0; half < 2; ++half) {
      float4 fk[4];
#pragma unroll
      for (int j = 0; j < 4; ++j) {
        const int i = tid + (half * 4 + j) * 128;
        const int r = i >> 4, c = (i & 15) << 2;
        fk[j] = *(const float4*)&k[kvb + (size_t)(kb + r) * DK_ + c];
      }
#pragma unroll
      for (int j = 0; j < 4; ++j) {
        const int i = tid + (half * 4 + j) * 128;
        const int r = i >> 4, c = (i & 15) << 2;
        *(v4h*)&Ks[r][c] = cvt4(fk[j]);
      }
    }
    // mask tile (int4 -> packed bytes), batched
#pragma unroll
    for (int half = 0; half < 2; ++half) {
      int4 mi[4];
#pragma unroll
      for (int j = 0; j < 4; ++j) {
        const int i = tid + (half * 4 + j) * 128;
        const int r = i >> 4, c = (i & 15) << 2;
        mi[j] = *(const int4*)&mask[mb + (size_t)(qb + r) * S_ + (kb + c)];
      }
#pragma unroll
      for (int j = 0; j < 4; ++j) {
        const int i = tid + (half * 4 + j) * 128;
        const int r = i >> 4, c = (i & 15) << 2;
        const unsigned int pk =
            (mi[j].x ? 1u : 0u) | ((mi[j].y ? 1u : 0u) << 8) |
            ((mi[j].z ? 1u : 0u) << 16) | ((mi[j].w ? 1u : 0u) << 24);
        *(unsigned int*)&Ms[r][c] = pk;
      }
    }
    // V tile, K-transposed to [dk][key]: 4-deep key columns, batched
#pragma unroll
    for (int half = 0; half < 2; ++half) {
      float fv[16];
#pragma unroll
      for (int jq = 0; jq < 4; ++jq) {
        const int i = tid + (half * 4 + jq) * 128;
        const int n = i & 63, k4 = (i >> 6) << 2;
#pragma unroll
        for (int j = 0; j < 4; ++j)
          fv[jq * 4 + j] = v[kvb + (size_t)(kb + k4 + j) * DK_ + n];
      }
#pragma unroll
      for (int jq = 0; jq < 4; ++jq) {
        const int i = tid + (half * 4 + jq) * 128;
        const int n = i & 63, k4 = (i >> 6) << 2;
        v4h hv;
#pragma unroll
        for (int j = 0; j < 4; ++j) hv[j] = (_Float16)fv[jq * 4 + j];
        *(v4h*)&VsT[n][k4] = hv;
      }
    }
    // prefetch next K/V tiles toward L2 while this block computes
    if (kb + 64 < S_) {
      const size_t nr = kvb + (size_t)(kb + 64 + (tid >> 1)) * DK_ + (tid & 1) * 32;
      __builtin_prefetch(&k[nr], 0, 0);
      __builtin_prefetch(&v[nr], 0, 0);
    }
    __syncthreads();

    // S = Q K^T  (reduction over d_k = 64, two K=32 chunks)
    v8f sc[4] = {};
#pragma unroll
    for (int kk = 0; kk < DK_; kk += 32) {
      const v16h af = load_frag_a(&Qs[wave * 16][kk], T_LD);
#pragma unroll
      for (int t = 0; t < 4; ++t) {
        const v16h bf = load_frag_bt(&Ks[t * 16][kk], T_LD);
        sc[t] = __builtin_amdgcn_wmma_f32_16x16x32_f16(
            false, af, false, bf, (short)0, sc[t], false, false);
      }
    }

    // mask + online softmax (row lives in one half-wave)
#pragma unroll
    for (int e = 0; e < 8; ++e) {
      const int rl = wave * 16 + mh + e;   // row within tile
      float mx = -1e30f;
#pragma unroll
      for (int t = 0; t < 4; ++t) {
        float s = sc[t][e];
        if (!Ms[rl][t * 16 + n0]) s = -1e9f;
        sc[t][e] = s;
        mx = fmaxf(mx, s);
      }
#pragma unroll
      for (int d = 1; d < 16; d <<= 1) mx = fmaxf(mx, __shfl_xor(mx, d, 32));
      const float mnew = fmaxf(mrun[e], mx);
      float sum = 0.0f;
#pragma unroll
      for (int t = 0; t < 4; ++t) {
        const float p = __expf(sc[t][e] - mnew);
        sc[t][e] = p;
        sum += p;
      }
#pragma unroll
      for (int d = 1; d < 16; d <<= 1) sum += __shfl_xor(sum, d, 32);
      const float resc = __expf(mrun[e] - mnew);
      lrun[e] = lrun[e] * resc + sum;
      mrun[e] = mnew;
#pragma unroll
      for (int t = 0; t < 4; ++t) o[t][e] *= resc;
#pragma unroll
      for (int t = 0; t < 4; ++t)
        Ps[rl][t * 16 + n0] = (_Float16)sc[t][e];
    }
    __syncthreads();

    // O += P V  (reduction over 64 keys, two K=32 chunks)
#pragma unroll
    for (int kk = 0; kk < 64; kk += 32) {
      const v16h af = load_frag_a(&Ps[wave * 16][kk], T_LD);
#pragma unroll
      for (int t = 0; t < 4; ++t) {
        const v16h bf = load_frag_bt(&VsT[t * 16][kk], T_LD);
        o[t] = __builtin_amdgcn_wmma_f32_16x16x32_f16(
            false, af, false, bf, (short)0, o[t], false, false);
      }
    }
    __syncthreads();
  }

  const size_t xb = (size_t)b * S_ * DM_;
#pragma unroll
  for (int t = 0; t < 4; ++t)
#pragma unroll
    for (int e = 0; e < 8; ++e) {
      const int row = qb + wave * 16 + mh + e;
      const int col = h * DK_ + t * 16 + n0;
      x[xb + (size_t)row * DM_ + col] = o[t][e] / lrun[e];
    }
}

// ---------------------------------------------------------------------------
extern "C" void kernel_launch(void* const* d_in, const int* in_sizes, int n_in,
                              void* d_out, int out_size, void* d_ws, size_t ws_size,
                              hipStream_t stream) {
  (void)in_sizes; (void)n_in; (void)out_size; (void)ws_size;
  const float* query = (const float*)d_in[0];
  const float* key   = (const float*)d_in[1];
  const float* value = (const float*)d_in[2];
  const int*   mask  = (const int*)d_in[3];
  const float* Wq = (const float*)d_in[4];
  const float* bq = (const float*)d_in[5];
  const float* Wk = (const float*)d_in[6];
  const float* bk = (const float*)d_in[7];
  const float* Wv = (const float*)d_in[8];
  const float* bv = (const float*)d_in[9];
  const float* Wo = (const float*)d_in[10];
  const float* bo = (const float*)d_in[11];
  float* out = (float*)d_out;

  const int M = B_ * S_;  // 4096
  float* qws = (float*)d_ws;                 // [M, 1024]
  float* kws = qws + (size_t)M * DM_;        // [M, 64]
  float* vws = kws + (size_t)M * DK_;        // [M, 64]
  float* xws = vws + (size_t)M * DK_;        // [M, 1024]

  dim3 blk(128);
  gemm_bias_f16wmma<<<dim3(DM_ / 64, M / 64), blk, 0, stream>>>(query, Wq, bq, qws, M, DM_, DM_);
  gemm_bias_f16wmma<<<dim3(DK_ / 64, M / 64), blk, 0, stream>>>(key,   Wk, bk, kws, M, DK_, DM_);
  gemm_bias_f16wmma<<<dim3(DK_ / 64, M / 64), blk, 0, stream>>>(value, Wv, bv, vws, M, DK_, DM_);
  flash_mqa<<<dim3(S_ / 64, B_ * H_), blk, 0, stream>>>(qws, kws, vws, mask, xws);
  gemm_bias_f16wmma<<<dim3(DM_ / 64, M / 64), blk, 0, stream>>>(xws, Wo, bo, out, M, DM_, DM_);
}
